// GraphSAGE_12850542150068
// MI455X (gfx1250) — matrix operand order repaired and checked by reading, not compiled, not run on hardware
//
#include <hip/hip_runtime.h>

typedef __attribute__((ext_vector_type(16))) __bf16 v16bf;
typedef __attribute__((ext_vector_type(8)))  __bf16 v8bf;
typedef __attribute__((ext_vector_type(8)))  float  v8f;

#define NNODES 50000
#define NEDGES 800000
#define DIN    128
#define DHID   256

__device__ __forceinline__ void fadd_atomic(float* p, float v) {
    // Native global_atomic_add_f32 (coarse-grained memory, no ordering needed).
    unsafeAtomicAdd(p, v);
}

// ---------------- degree ----------------
__global__ void deg_count_kernel(const long long* __restrict__ dst, float* __restrict__ deg, int e) {
    int t = blockIdx.x * blockDim.x + threadIdx.x;
    if (t < e) fadd_atomic(&deg[(int)dst[t]], 1.0f);
}

__global__ void deg_inv_kernel(float* __restrict__ deg, int n) {
    int t = blockIdx.x * blockDim.x + threadIdx.x;
    if (t < n) { float d = deg[t]; deg[t] = (d > 0.0f) ? (1.0f / d) : 0.0f; }
}

// ---------------- scatter-add (gather float4, hw fp32 atomics) ----------------
// Consecutive threads cover one edge's feature row contiguously -> coalesced
// 512B/1KB gathers; atomics are native fp32 adds to the dst row.
__global__ void scatter_add_kernel(const float* __restrict__ in,
                                   const long long* __restrict__ src,
                                   const long long* __restrict__ dst,
                                   float* __restrict__ agg, int e, int d) {
    long long t = (long long)blockIdx.x * blockDim.x + threadIdx.x;
    int chunks = d >> 2;
    long long total = (long long)e * chunks;
    if (t >= total) return;
    int ei = (int)(t / chunks);
    int c  = (int)(t % chunks) << 2;
    int s  = (int)src[ei];
    int dd = (int)dst[ei];
    float4 v = *(const float4*)(in + (size_t)s * d + c);
    float* p = agg + (size_t)dd * d + c;
    fadd_atomic(p + 0, v.x);
    fadd_atomic(p + 1, v.y);
    fadd_atomic(p + 2, v.z);
    fadd_atomic(p + 3, v.w);
}

// ---------------- pack A = [agg*deg_inv | root] as bf16, ld = 2*d ----------------
__global__ void pack_pair_kernel(const float* __restrict__ agg,
                                 const float* __restrict__ root,
                                 const float* __restrict__ dinv,
                                 __bf16* __restrict__ A, int n, int d) {
    long long t = (long long)blockIdx.x * blockDim.x + threadIdx.x;
    long long total = (long long)n * d;
    if (t >= total) return;
    int i = (int)(t / d), c = (int)(t % d);
    float s = dinv[i];
    size_t base = (size_t)i * (2 * d);
    A[base + c]     = (__bf16)(agg[(size_t)i * d + c] * s);
    A[base + d + c] = (__bf16)(root[(size_t)i * d + c]);
}

// ---------------- weights: Wt[n, k] = (k<K ? Wl[k,n] : Wr[k-K,n]) col-major bf16 ----------------
__global__ void prep_wt_kernel(const float* __restrict__ Wl, const float* __restrict__ Wr,
                               __bf16* __restrict__ Wt, int K, int dout) {
    int t = blockIdx.x * blockDim.x + threadIdx.x;
    int K2 = 2 * K;
    if (t >= dout * K2) return;
    int nn = t / K2, k = t % K2;
    float v = (k < K) ? Wl[(size_t)k * dout + nn] : Wr[(size_t)(k - K) * dout + nn];
    Wt[(size_t)nn * K2 + k] = (__bf16)v;
}

// ---------------- WMMA bf16 GEMM + bias + ReLU ----------------
// OUT[n x 256] = relu(A[n x K](bf16) @ Wt^T + bias), Wt col-major (256 x K) bf16.
// 256 threads = 8 waves; each wave computes a 16x64 D tile (4 accumulators share
// one A fragment per K-step); fragments double-buffered across K-steps.
template<int K, bool RELU>
__global__ __launch_bounds__(256) void gemm_bias_act_kernel(
        const __bf16* __restrict__ A, const __bf16* __restrict__ Wt,
        const float* __restrict__ bias, float* __restrict__ out, int nrows) {
    const int lane = threadIdx.x & 31;
    const int wave = threadIdx.x >> 5;
    const int half = lane >> 4;   // 0: lanes 0-15, 1: lanes 16-31
    const int l16  = lane & 15;
    const int m0 = blockIdx.x * 128 + wave * 16;
    const int n0 = blockIdx.y * 64;

    int arow = m0 + l16;
    if (arow >= nrows) arow = nrows - 1;       // clamp loads; stores are guarded
    const __bf16* Aptr = A + (size_t)arow * K + half * 16;
    const __bf16* Bp[4] = {
        Wt + (size_t)(n0 +  0 + l16) * K + half * 16,
        Wt + (size_t)(n0 + 16 + l16) * K + half * 16,
        Wt + (size_t)(n0 + 32 + l16) * K + half * 16,
        Wt + (size_t)(n0 + 48 + l16) * K + half * 16,
    };

    // A frag: lanes<16 hold K={k0..k0+15}; lanes>=16 shifted by +16
    auto load_a = [&](int k0) {
        v8bf alo = *(const v8bf*)(Aptr + k0);
        v8bf ahi = *(const v8bf*)(Aptr + k0 + 8);
        v16bf a;
#pragma unroll
        for (int e = 0; e < 8; ++e) { a[e] = alo[e]; a[e + 8] = ahi[e]; }
        return a;
    };

    v8f c0 = {}, c1 = {}, c2 = {}, c3 = {};

    v16bf a_cur = load_a(0);
    v16bf b_cur0 = *(const v16bf*)(Bp[0]);
    v16bf b_cur1 = *(const v16bf*)(Bp[1]);
    v16bf b_cur2 = *(const v16bf*)(Bp[2]);
    v16bf b_cur3 = *(const v16bf*)(Bp[3]);

#pragma unroll
    for (int k0 = 0; k0 < K; k0 += 32) {
        const int kn = k0 + 32;
        v16bf a_nxt, b_nxt0, b_nxt1, b_nxt2, b_nxt3;
        if (kn < K) {                           // issue next-step loads first
            a_nxt  = load_a(kn);
            b_nxt0 = *(const v16bf*)(Bp[0] + kn);
            b_nxt1 = *(const v16bf*)(Bp[1] + kn);
            b_nxt2 = *(const v16bf*)(Bp[2] + kn);
            b_nxt3 = *(const v16bf*)(Bp[3] + kn);
        }
        c0 = __builtin_amdgcn_wmma_f32_16x16x32_bf16(false, a_cur, false, b_cur0, (short)0, c0, false, false);
        c1 = __builtin_amdgcn_wmma_f32_16x16x32_bf16(false, a_cur, false, b_cur1, (short)0, c1, false, false);
        c2 = __builtin_amdgcn_wmma_f32_16x16x32_bf16(false, a_cur, false, b_cur2, (short)0, c2, false, false);
        c3 = __builtin_amdgcn_wmma_f32_16x16x32_bf16(false, a_cur, false, b_cur3, (short)0, c3, false, false);
        if (kn < K) {
            a_cur = a_nxt;
            b_cur0 = b_nxt0; b_cur1 = b_nxt1; b_cur2 = b_nxt2; b_cur3 = b_nxt3;
        }
    }

    v8f acc[4] = {c0, c1, c2, c3};
#pragma unroll
    for (int j = 0; j < 4; ++j) {
        const int col = n0 + j * 16 + l16;
        const float bv = bias[col];
#pragma unroll
        for (int r = 0; r < 8; ++r) {
            int row = m0 + r + half * 8;       // VGPR r: M = r (lanes<16) / r+8 (lanes>=16)
            if (row < nrows) {
                float v = acc[j][r] + bv;
                if (RELU) v = v > 0.0f ? v : 0.0f;
                out[(size_t)row * 256 + col] = v;
            }
        }
    }
}

// ---------------- classifier: out[n x 2] = h @ W3 + b3, one wave per node ----------------
__global__ void classifier_kernel(const float* __restrict__ h, const float* __restrict__ W3,
                                  const float* __restrict__ b3, float* __restrict__ out, int n) {
    long long g = (long long)blockIdx.x * blockDim.x + threadIdx.x;
    int node = (int)(g >> 5);
    int lane = (int)(g & 31);
    if (node >= n) return;
    const float* hr = h + (size_t)node * 256;
    float a0 = 0.0f, a1 = 0.0f;
#pragma unroll
    for (int j = 0; j < 8; ++j) {
        int k = lane + j * 32;
        float v = hr[k];
        a0 += v * W3[k * 2 + 0];
        a1 += v * W3[k * 2 + 1];
    }
#pragma unroll
    for (int off = 16; off > 0; off >>= 1) {
        a0 += __shfl_down(a0, off, 32);
        a1 += __shfl_down(a1, off, 32);
    }
    if (lane == 0) {
        out[(size_t)node * 2 + 0] = a0 + b3[0];
        out[(size_t)node * 2 + 1] = a1 + b3[1];
    }
}

extern "C" void kernel_launch(void* const* d_in, const int* in_sizes, int n_in,
                              void* d_out, int out_size, void* d_ws, size_t ws_size,
                              hipStream_t stream) {
    const float*     x    = (const float*)d_in[0];
    const long long* ei   = (const long long*)d_in[1];
    const float*     W1l  = (const float*)d_in[2];
    const float*     b1   = (const float*)d_in[3];
    const float*     W1r  = (const float*)d_in[4];
    const float*     W2l  = (const float*)d_in[5];
    const float*     b2   = (const float*)d_in[6];
    const float*     W2r  = (const float*)d_in[7];
    const float*     W3   = (const float*)d_in[8];
    const float*     b3   = (const float*)d_in[9];
    const long long* src  = ei;
    const long long* dst  = ei + NEDGES;

    // workspace bump allocator (256B aligned)
    size_t off = 0;
    auto alloc = [&](size_t bytes) -> void* {
        off = (off + 255) & ~(size_t)255;
        void* p = (char*)d_ws + off;
        off += bytes;
        return p;
    };
    float*  deg = (float*)alloc((size_t)NNODES * 4);                 // becomes deg_inv
    __bf16* A   = (__bf16*)alloc((size_t)NNODES * (2 * DHID) * 2);   // A1 (ld 256) then A2 (ld 512)
    float*  agg = (float*)alloc((size_t)NNODES * DHID * 4);          // agg1/agg2, then h2
    float*  h1  = (float*)alloc((size_t)NNODES * DHID * 4);
    __bf16* Wt1 = (__bf16*)alloc((size_t)DHID * (2 * DIN) * 2);      // 256 x 256
    __bf16* Wt2 = (__bf16*)alloc((size_t)DHID * (2 * DHID) * 2);     // 256 x 512

    const int TB = 256;
    // ---- degree / mean scale ----
    hipMemsetAsync(deg, 0, (size_t)NNODES * 4, stream);
    deg_count_kernel<<<(NEDGES + TB - 1) / TB, TB, 0, stream>>>(dst, deg, NEDGES);
    deg_inv_kernel<<<(NNODES + TB - 1) / TB, TB, 0, stream>>>(deg, NNODES);

    // ---- layer 1 ----
    hipMemsetAsync(agg, 0, (size_t)NNODES * DIN * 4, stream);
    {
        long long tot = (long long)NEDGES * (DIN / 4);
        scatter_add_kernel<<<(unsigned)((tot + TB - 1) / TB), TB, 0, stream>>>(x, src, dst, agg, NEDGES, DIN);
    }
    {
        long long tot = (long long)NNODES * DIN;
        pack_pair_kernel<<<(unsigned)((tot + TB - 1) / TB), TB, 0, stream>>>(agg, x, deg, A, NNODES, DIN);
    }
    prep_wt_kernel<<<(DHID * 2 * DIN + TB - 1) / TB, TB, 0, stream>>>(W1l, W1r, Wt1, DIN, DHID);
    {
        dim3 grid((NNODES + 127) / 128, DHID / 64);
        gemm_bias_act_kernel<2 * DIN, true><<<grid, TB, 0, stream>>>(A, Wt1, b1, h1, NNODES);
    }

    // ---- layer 2 ----
    hipMemsetAsync(agg, 0, (size_t)NNODES * DHID * 4, stream);
    {
        long long tot = (long long)NEDGES * (DHID / 4);
        scatter_add_kernel<<<(unsigned)((tot + TB - 1) / TB), TB, 0, stream>>>(h1, src, dst, agg, NEDGES, DHID);
    }
    {
        long long tot = (long long)NNODES * DHID;
        pack_pair_kernel<<<(unsigned)((tot + TB - 1) / TB), TB, 0, stream>>>(agg, h1, deg, A, NNODES, DHID);
    }
    prep_wt_kernel<<<(DHID * 2 * DHID + TB - 1) / TB, TB, 0, stream>>>(W2l, W2r, Wt2, DHID, DHID);
    float* h2 = agg;  // agg dead after A2 built; reuse for h2
    {
        dim3 grid((NNODES + 127) / 128, DHID / 64);
        gemm_bias_act_kernel<2 * DHID, true><<<grid, TB, 0, stream>>>(A, Wt2, b2, h2, NNODES);
    }

    // ---- classifier ----
    {
        long long tot = (long long)NNODES * 32;
        classifier_kernel<<<(unsigned)((tot + TB - 1) / TB), TB, 0, stream>>>(h2, W3, b3, (float*)d_out, NNODES);
    }
}
